// MultiHeadAttention_50199577755728
// MI455X (gfx1250) — matrix operand order repaired and checked by reading, not compiled, not run on hardware
//
#include <hip/hip_runtime.h>
#include <cstdint>
#include <cstddef>

// Problem constants (match reference): B=4, S=2048, E=1024, H=16, DH=64, NI=4096
#define Bn  4
#define Sn  2048
#define En  1024
#define Hn  16
#define DHn 64
#define NIn 4096

typedef __bf16 bf16;
typedef __attribute__((ext_vector_type(16))) __bf16        v16bf;
typedef __attribute__((ext_vector_type(8)))  float         v8f;
typedef __attribute__((ext_vector_type(4)))  unsigned int  u32x4;

union Frag16 { v16bf v; u32x4 q[2]; };

__device__ __forceinline__ bf16 f2bf(float f) { return (bf16)f; }

__device__ __forceinline__ v8f vzero8() {
  v8f z = {0.f, 0.f, 0.f, 0.f, 0.f, 0.f, 0.f, 0.f};
  return z;
}

__device__ __forceinline__ v8f wmma_bf16(const Frag16& a, const Frag16& b, v8f c) {
  // D = A(16x32 bf16) x B(32x16 bf16) + C(16x16 f32)
  return __builtin_amdgcn_wmma_f32_16x16x32_bf16(
      /*neg_a=*/false, a.v, /*neg_b=*/false, b.v,
      /*c_mod=*/(short)0, c, /*reuse_a=*/false, /*reuse_b=*/false);
}

// Async global->LDS copy (16B per lane), tracked by ASYNCcnt. Bypasses VGPRs.
__device__ __forceinline__ void async_g2l_b128(unsigned lds_byte_off, const void* gaddr) {
  asm volatile("global_load_async_to_lds_b128 %0, %1, off"
               :
               : "v"(lds_byte_off), "v"(gaddr)
               : "memory");
}

// ---------------------------------------------------------------------------
// Conversion kernels: fp32 -> bf16, weights transposed to [N][K] layout so
// GEMM B-tiles are contiguous along K (matches WMMA B-fragment lane layout).
// ---------------------------------------------------------------------------
__global__ void cast_f32_to_bf16(const float* __restrict__ in, bf16* __restrict__ out, int n) {
  int i = blockIdx.x * blockDim.x + threadIdx.x;
  if (i < n) out[i] = f2bf(in[i]);
}

// Wq/Wk/Wv [H, E, DH] -> WT [N=H*DH][K=E]:  WT[(h*64+d)*E + e] = W[(h*E+e)*DH + d]
__global__ void conv_wqkv_T(const float* __restrict__ w, bf16* __restrict__ wt) {
  int i = blockIdx.x * blockDim.x + threadIdx.x;     // over 1024*E
  if (i >= Hn * DHn * En) return;
  int e = i % En;
  int n = i / En;
  int h = n >> 6, d = n & 63;
  wt[i] = f2bf(w[(h * En + e) * DHn + d]);
}

// W1 [E, NI] -> W1T [N=NI][K=E]
__global__ void conv_w1_T(const float* __restrict__ w, bf16* __restrict__ wt) {
  int i = blockIdx.x * blockDim.x + threadIdx.x;     // over NI*E
  if (i >= NIn * En) return;
  int e = i % En;
  int n = i / En;
  wt[i] = f2bf(w[e * NIn + n]);
}

// W2 [NI, E] -> W2T [N=E][K=NI]
__global__ void conv_w2_T(const float* __restrict__ w, bf16* __restrict__ wt) {
  int i = blockIdx.x * blockDim.x + threadIdx.x;     // over E*NI
  if (i >= En * NIn) return;
  int k = i % NIn;
  int n = i / NIn;
  wt[i] = f2bf(w[k * En + n]);
}

// ---------------------------------------------------------------------------
// bf16 GEMM: C[M,N] = A[M,K] (row-major) * BT[N,K] (B stored col-major).
// Block tile 256x128, K-step 32, double-buffered LDS filled by
// global_load_async_to_lds_b128 (ASYNCcnt). 8 waves as 4x2; each wave owns a
// 64x64 tile = 4x4 WMMA tiles -> 16 WMMAs per 16 ds_load_b128 fragment reads.
// EPI: 0 = store bf16; 1 = +bias, ReLU, store bf16; 2 = +bias +residual, f32.
// ---------------------------------------------------------------------------
template <int EPI>
__global__ __launch_bounds__(256)
void gemm_bf16_wmma(const bf16* __restrict__ A, const bf16* __restrict__ BT,
                    void* __restrict__ Cout, const float* __restrict__ bias,
                    const float* __restrict__ resid, int M, int N, int K) {
  __shared__ __align__(16) bf16 As[2][256][40];   // padded: 80B row stride
  __shared__ __align__(16) bf16 Bs[2][128][40];   // Bs[n][k]

  const int tid  = threadIdx.x;
  const int wid  = tid >> 5;
  const int lane = tid & 31;
  const int l15  = lane & 15;
  const int half = lane >> 4;
  const int m0 = blockIdx.y * 256;
  const int n0 = blockIdx.x * 128;
  const int wm = (wid >> 1) * 64;   // wave row offset in block
  const int wn = (wid & 1) * 64;    // wave col offset in block

  // LDS byte offsets (generic-shared address low 32 bits == LDS offset).
  const unsigned asBase = (unsigned)(uintptr_t)(void*)As;
  const unsigned bsBase = (unsigned)(uintptr_t)(void*)Bs;

  v8f acc[4][4];
#pragma unroll
  for (int mt = 0; mt < 4; mt++)
#pragma unroll
    for (int nt = 0; nt < 4; nt++) acc[mt][nt] = vzero8();

  // Issue one tile's async loads: A 256x32 (4 b128/thread), B 128x32 (2/thread).
  auto issue_tile = [&](int buf, int k0) {
#pragma unroll
    for (int i = 0; i < 4; i++) {
      int c = tid + i * 256;          // 0..1023
      int r = c >> 2;                 // 0..255
      int col = (c & 3) * 8;          // 0,8,16,24
      async_g2l_b128(asBase + (((unsigned)buf * 256 + r) * 40 + col) * 2,
                     A + (size_t)(m0 + r) * K + k0 + col);
    }
#pragma unroll
    for (int i = 0; i < 2; i++) {
      int c = tid + i * 256;          // 0..511
      int r = c >> 2;                 // 0..127
      int col = (c & 3) * 8;
      async_g2l_b128(bsBase + (((unsigned)buf * 128 + r) * 40 + col) * 2,
                     BT + (size_t)(n0 + r) * K + k0 + col);
    }
  };

  issue_tile(0, 0);
  int cur = 0;
  for (int k0 = 0; k0 < K; k0 += 32) {
    if (k0 + 32 < K) {
      issue_tile(cur ^ 1, k0 + 32);                      // prefetch next tile
      asm volatile("s_wait_asynccnt 0x6" ::: "memory");  // current tile landed
    } else {
      asm volatile("s_wait_asynccnt 0x0" ::: "memory");
    }
    __syncthreads();

    // A fragments: lane = row m, K pairs at [8h,8h+8) and [16+8h,16+8h+8)
    Frag16 af[4];
#pragma unroll
    for (int mt = 0; mt < 4; mt++) {
      const int m = wm + mt * 16 + l15;
      af[mt].q[0] = *(const u32x4*)(&As[cur][m][half * 8]);
      af[mt].q[1] = *(const u32x4*)(&As[cur][m][16 + half * 8]);
    }
    // B fragments: lane = column n, K at [8h,8h+8) and [16+8h,16+8h+8)
    Frag16 bfr[4];
#pragma unroll
    for (int nt = 0; nt < 4; nt++) {
      const int n = wn + nt * 16 + l15;
      bfr[nt].q[0] = *(const u32x4*)(&Bs[cur][n][half * 8]);
      bfr[nt].q[1] = *(const u32x4*)(&Bs[cur][n][16 + half * 8]);
    }
#pragma unroll
    for (int mt = 0; mt < 4; mt++)
#pragma unroll
      for (int nt = 0; nt < 4; nt++)
        acc[mt][nt] = wmma_bf16(af[mt], bfr[nt], acc[mt][nt]);

    __syncthreads();   // everyone done reading buf `cur` before it is refilled
    cur ^= 1;
  }

  // Epilogue. C-frag layout: VGPR r -> row r + 8*half, lane&15 -> col.
#pragma unroll
  for (int mt = 0; mt < 4; mt++)
#pragma unroll
    for (int nt = 0; nt < 4; nt++)
#pragma unroll
      for (int r = 0; r < 8; r++) {
        int gm = m0 + wm + mt * 16 + r + 8 * half;
        int gn = n0 + wn + nt * 16 + l15;
        float v = acc[mt][nt][r];
        if (EPI == 1) { v += bias[gn]; v = fmaxf(v, 0.0f); }
        if (EPI == 2) { v += bias[gn] + resid[(size_t)gm * N + gn]; }
        if (EPI == 2) ((float*)Cout)[(size_t)gm * N + gn] = v;
        else          ((bf16*)Cout)[(size_t)gm * N + gn] = f2bf(v);
      }
}

// ---------------------------------------------------------------------------
// Flash attention with WMMA. One block = (b, h, 128 query rows); 8 waves,
// each wave owns 16 query rows with the full 16x64 O accumulator.
// Q/K/V stored bf16 as [B*S, E] with head h at column h*64 (concat layout).
// ---------------------------------------------------------------------------
__global__ __launch_bounds__(256)
void attn_flash_wmma(const bf16* __restrict__ Q, const bf16* __restrict__ Kb,
                     const bf16* __restrict__ Vb, bf16* __restrict__ Y) {
  __shared__ __align__(16) bf16 VsT[64][40];     // V chunk transposed: [d][key]
  __shared__ __align__(16) bf16 Ps[8][16][40];   // per-wave P tile (16x32)

  const int tid  = threadIdx.x;
  const int wid  = tid >> 5;
  const int lane = tid & 31;
  const int l15  = lane & 15;
  const int half = lane >> 4;

  int bx = blockIdx.x;
  const int qb = bx & 15; bx >>= 4;    // S/128 = 16 query blocks
  const int h  = bx & 15; bx >>= 4;    // H = 16
  const int b  = bx;                   // B = 4
  const int q0 = qb * 128;
  const int wq = q0 + wid * 16;        // this wave's first query row

  const size_t basebh = (size_t)b * Sn * En + (size_t)h * DHn;

  // Q fragments (persistent): 2 K-steps covering DH=64.
  Frag16 qf[2];
  {
    const bf16* qrow = Q + basebh + (size_t)(wq + l15) * En;
#pragma unroll
    for (int kk = 0; kk < 2; kk++) {
      qf[kk].q[0] = *(const u32x4*)(qrow + kk * 32 + half * 8);
      qf[kk].q[1] = *(const u32x4*)(qrow + kk * 32 + 16 + half * 8);
    }
  }

  v8f o[4];
#pragma unroll
  for (int t = 0; t < 4; t++) o[t] = vzero8();
  float mrow[8], lrow[8];
#pragma unroll
  for (int r = 0; r < 8; r++) { mrow[r] = -1e30f; lrow[r] = 0.0f; }

  const int nchunk = (q0 + 128) / 32;   // causal: keys up to block's last row
  for (int c = 0; c < nchunk; c++) {
    const int t0 = c * 32;
    __syncthreads();  // previous iteration done reading VsT
    {   // stage V chunk (32 keys x 64 d) transposed into LDS
      int key   = tid >> 3;
      int dbase = (tid & 7) * 8;
      u32x4 vv = *(const u32x4*)(Vb + basebh + (size_t)(t0 + key) * En + dbase);
      const bf16* pv = (const bf16*)&vv;
#pragma unroll
      for (int i = 0; i < 8; i++) VsT[dbase + i][key] = pv[i];
    }
    __syncthreads();

    // Scores S = Q * K^T : K rows in global are exactly the B-fragment layout.
    v8f s[2];
#pragma unroll
    for (int nt = 0; nt < 2; nt++) {
      s[nt] = vzero8();
      const bf16* krow = Kb + basebh + (size_t)(t0 + nt * 16 + l15) * En;
#pragma unroll
      for (int kk = 0; kk < 2; kk++) {
        Frag16 kf;
        kf.q[0] = *(const u32x4*)(krow + kk * 32 + half * 8);
        kf.q[1] = *(const u32x4*)(krow + kk * 32 + 16 + half * 8);
        s[nt] = wmma_bf16(qf[kk], kf, s[nt]);
      }
    }
    // Scale + causal mask (finite mask value keeps exp() NaN-free).
#pragma unroll
    for (int nt = 0; nt < 2; nt++)
#pragma unroll
      for (int r = 0; r < 8; r++) {
        int q = wq + r + 8 * half;
        int t = t0 + nt * 16 + l15;
        float v = s[nt][r] * 0.125f;   // 1/sqrt(64)
        s[nt][r] = (t <= q) ? v : -1e30f;
      }

    // Online softmax per row (row = one lane-half of a VGPR).
#pragma unroll
    for (int r = 0; r < 8; r++) {
      float rm = fmaxf(s[0][r], s[1][r]);
#pragma unroll
      for (int off = 8; off >= 1; off >>= 1) rm = fmaxf(rm, __shfl_xor(rm, off, 32));
      float mo = mrow[r];
      float mn = fmaxf(mo, rm);
      float alpha = __expf(mo - mn);
      float p0 = __expf(s[0][r] - mn);
      float p1 = __expf(s[1][r] - mn);
      float ps = p0 + p1;
#pragma unroll
      for (int off = 8; off >= 1; off >>= 1) ps += __shfl_xor(ps, off, 32);
      lrow[r] = lrow[r] * alpha + ps;
      mrow[r] = mn;
#pragma unroll
      for (int t = 0; t < 4; t++) o[t][r] *= alpha;
      int prow = r + 8 * half;
      Ps[wid][prow][l15]      = f2bf(p0);
      Ps[wid][prow][16 + l15] = f2bf(p1);
    }
    __syncthreads();  // make Ps (C-layout stores) visible to A-layout loads

    // O += P(16x32) * V(32x64)
    Frag16 pf;
    pf.q[0] = *(const u32x4*)(&Ps[wid][l15][half * 8]);
    pf.q[1] = *(const u32x4*)(&Ps[wid][l15][16 + half * 8]);
#pragma unroll
    for (int t = 0; t < 4; t++) {
      Frag16 vf;
      int d = t * 16 + l15;
      vf.q[0] = *(const u32x4*)(&VsT[d][half * 8]);
      vf.q[1] = *(const u32x4*)(&VsT[d][16 + half * 8]);
      o[t] = wmma_bf16(pf, vf, o[t]);
    }
  }

  // Finalize: O / l, write concat-head layout [B*S, E], e = h*64 + d.
  float inv[8];
#pragma unroll
  for (int r = 0; r < 8; r++) inv[r] = 1.0f / lrow[r];
#pragma unroll
  for (int t = 0; t < 4; t++)
#pragma unroll
    for (int r = 0; r < 8; r++) {
      int q = wq + r + 8 * half;
      int e = h * DHn + t * 16 + l15;
      Y[((size_t)b * Sn + q) * En + e] = f2bf(o[t][r] * inv[r]);
    }
}

// ---------------------------------------------------------------------------
extern "C" void kernel_launch(void* const* d_in, const int* in_sizes, int n_in,
                              void* d_out, int out_size, void* d_ws, size_t ws_size,
                              hipStream_t stream) {
  const float* x  = (const float*)d_in[0];
  const float* Wq = (const float*)d_in[1];
  const float* Wk = (const float*)d_in[2];
  const float* Wv = (const float*)d_in[3];
  const float* W1 = (const float*)d_in[4];
  const float* b1 = (const float*)d_in[5];
  const float* W2 = (const float*)d_in[6];
  const float* b2 = (const float*)d_in[7];
  float* out = (float*)d_out;

  const size_t BSE = (size_t)Bn * Sn * En;   // 8,388,608
  const size_t HDE = (size_t)Hn * DHn * En;  // 1,048,576
  const size_t NIE = (size_t)NIn * En;       // 4,194,304
  const size_t BSNI = (size_t)Bn * Sn * NIn; // 33,554,432

  char* ws = (char*)d_ws;
  size_t off = 0;
  auto salloc = [&](size_t bytes) -> void* {
    void* p = ws + off;
    off += (bytes + 255) & ~(size_t)255;
    return p;
  };
  bf16* xb  = (bf16*)salloc(BSE * 2);
  bf16* WqT = (bf16*)salloc(HDE * 2);
  bf16* WkT = (bf16*)salloc(HDE * 2);
  bf16* WvT = (bf16*)salloc(HDE * 2);
  bf16* W1T = (bf16*)salloc(NIE * 2);
  bf16* W2T = (bf16*)salloc(NIE * 2);
  bf16* Qb  = (bf16*)salloc(BSE * 2);
  bf16* Kbf = (bf16*)salloc(BSE * 2);
  bf16* Vbf = (bf16*)salloc(BSE * 2);
  bf16* Yb  = (bf16*)salloc(BSE * 2);
  bf16* Hb  = (bf16*)salloc(BSNI * 2);
  (void)ws_size; (void)in_sizes; (void)n_in; (void)out_size;

  const int T = 256;
  cast_f32_to_bf16<<<(int)((BSE + T - 1) / T), T, 0, stream>>>(x, xb, (int)BSE);
  conv_wqkv_T<<<(int)((HDE + T - 1) / T), T, 0, stream>>>(Wq, WqT);
  conv_wqkv_T<<<(int)((HDE + T - 1) / T), T, 0, stream>>>(Wk, WkT);
  conv_wqkv_T<<<(int)((HDE + T - 1) / T), T, 0, stream>>>(Wv, WvT);
  conv_w1_T<<<(int)((NIE + T - 1) / T), T, 0, stream>>>(W1, W1T);
  conv_w2_T<<<(int)((NIE + T - 1) / T), T, 0, stream>>>(W2, W2T);

  const int Mrows = Bn * Sn;                 // 8192
  dim3 gQKV(En / 128, Mrows / 256);          // (8, 32)
  gemm_bf16_wmma<0><<<gQKV, 256, 0, stream>>>(xb, WqT, Qb, nullptr, nullptr, Mrows, En, En);
  gemm_bf16_wmma<0><<<gQKV, 256, 0, stream>>>(xb, WkT, Kbf, nullptr, nullptr, Mrows, En, En);
  gemm_bf16_wmma<0><<<gQKV, 256, 0, stream>>>(xb, WvT, Vbf, nullptr, nullptr, Mrows, En, En);

  attn_flash_wmma<<<Bn * Hn * (Sn / 128), 256, 0, stream>>>(Qb, Kbf, Vbf, Yb);

  dim3 gF1(NIn / 128, Mrows / 256);          // (32, 32)
  gemm_bf16_wmma<1><<<gF1, 256, 0, stream>>>(Yb, W1T, Hb, b1, nullptr, Mrows, NIn, En);
  dim3 gF2(En / 128, Mrows / 256);           // (8, 32)
  gemm_bf16_wmma<2><<<gF2, 256, 0, stream>>>(Hb, W2T, out, b2, x, Mrows, En, NIn);
}
